// GNN_24137716203574
// MI455X (gfx1250) — matrix operand order repaired and checked by reading, not compiled, not run on hardware
//
#include <hip/hip_runtime.h>
#include <hip/hip_bf16.h>

typedef __attribute__((ext_vector_type(16))) _Float16 v16h;
typedef __attribute__((ext_vector_type(2)))  _Float16 h2v;
typedef __attribute__((ext_vector_type(8)))  float    v8f;
typedef int v4i __attribute__((vector_size(16)));                 // matches builtin param
typedef v4i __attribute__((address_space(1)))* gv4i_p;            // global AS pointer
typedef v4i __attribute__((address_space(3)))* lv4i_p;            // LDS AS pointer

#define N_USR   60000
#define N_MOV   20000
#define N_TOT   80000
#define F_USR   64
#define F_MOV   128
#define INC     256
#define HIDC    256
#define OUTC    128
#define N_EDGE  500000
#define BN_EPS  1e-5f

#if defined(__has_builtin)
#if __has_builtin(__builtin_amdgcn_global_load_async_to_lds_b128) && \
    __has_builtin(__builtin_amdgcn_s_wait_asynccnt)
#define HAVE_ASYNC_LDS 1
#endif
#endif

__device__ __forceinline__ void atomAddF(float* p, float v) {
    __hip_atomic_fetch_add(p, v, __ATOMIC_RELAXED, __HIP_MEMORY_SCOPE_AGENT);
}

// ---------------- f32 -> f16 conversion (streaming) ----------------
__global__ void cvt_f32_f16(const float* __restrict__ in, _Float16* __restrict__ out, int n) {
    int i = blockIdx.x * blockDim.x + threadIdx.x;
    int stride = gridDim.x * blockDim.x;
    for (; i < n; i += stride) out[i] = (_Float16)in[i];
}

// ---------------- WMMA GEMM: C[M,N] = A[M,K] @ B[K,N] (+bias) ----------------
// 256 threads = 8 waves; block tile 64x64; wave tile 16x32 (2 accumulators);
// K-step 32. B (weights) staged transposed for the full K strip once per block;
// A tile staged per K-step, via async-to-LDS DMA when available.
template<bool OUT_F16>
__global__ __launch_bounds__(256) void wmma_gemm(
    const _Float16* __restrict__ A, const _Float16* __restrict__ B,
    const float* __restrict__ bias,
    float* __restrict__ Cf, _Float16* __restrict__ Ch,
    int M, int N, int K)
{
    __shared__ _Float16 As[64][40];     // 64 rows x 32 k (pitch 40 -> 80B, 16B aligned)
    __shared__ _Float16 Bst[64][264];   // [col][k], k up to 256 (pitch 264 -> 16B aligned)

    const int tid  = threadIdx.x;
    const int lane = tid & 31;
    const int wave = tid >> 5;      // 0..7
    const int wr   = wave >> 1;     // 0..3  -> 16-row strip
    const int wc   = wave & 1;      // 0..1  -> 32-col strip
    const int half = lane >> 4;     // 0/1
    const int r    = lane & 15;

    const int m0 = blockIdx.x * 64;
    const int n0 = blockIdx.y * 64;

    // ---- stage B transposed: Bst[col][k] = B[k][n0+col], all K ----
    {
        const int col   = tid & 63;
        const int kbase = (tid >> 6) * 8;   // 0,8,16,24
        for (int kk = kbase; kk < K; kk += 32) {
            union { _Float16 h[8]; uint4 u; } tmp;
#pragma unroll
            for (int j = 0; j < 8; ++j)
                tmp.h[j] = B[(size_t)(kk + j) * N + n0 + col];
            *(uint4*)&Bst[col][kk] = tmp.u;
        }
    }

    v8f acc0 = {};
    v8f acc1 = {};

    const int ar  = tid >> 2;           // 0..63
    const int ac  = (tid & 3) * 8;      // 0,8,16,24
    const int arow_lds = wr * 16 + r;

    __syncthreads();                    // Bst ready (also fences first A stage)

    for (int k0 = 0; k0 < K; k0 += 32) {
        // ---- stage A tile 64x32 (16B per thread) ----
        {
            int row = m0 + ar;
#if HAVE_ASYNC_LDS
            if (row < M) {
                __builtin_amdgcn_global_load_async_to_lds_b128(
                    (gv4i_p)(A + (size_t)row * K + k0 + ac),
                    (lv4i_p)&As[ar][ac],
                    0, 0);
            } else {
                *(uint4*)&As[ar][ac] = make_uint4(0u, 0u, 0u, 0u);
            }
            __builtin_amdgcn_s_wait_asynccnt(0);
#else
            uint4 v = make_uint4(0u, 0u, 0u, 0u);
            if (row < M) v = *(const uint4*)(A + (size_t)row * K + k0 + ac);
            *(uint4*)&As[ar][ac] = v;
#endif
        }
        __syncthreads();

        // ---- build fragments (ISA 7.12.2 layouts) and multiply ----
        v16h afrag, bfrag0, bfrag1;
#pragma unroll
        for (int v = 0; v < 8; ++v) {
            int ka = (v >> 2) * 16 + half * 8 + (v & 3) * 2;
            h2v pa = *(const h2v*)&As[arow_lds][ka];
            afrag[2 * v]     = pa[0];
            afrag[2 * v + 1] = pa[1];
            int kb = k0 + (v >> 2) * 16 + half * 8 + (v & 3) * 2;
            h2v pb0 = *(const h2v*)&Bst[wc * 32 + r][kb];
            h2v pb1 = *(const h2v*)&Bst[wc * 32 + 16 + r][kb];
            bfrag0[2 * v]     = pb0[0];
            bfrag0[2 * v + 1] = pb0[1];
            bfrag1[2 * v]     = pb1[0];
            bfrag1[2 * v + 1] = pb1[1];
        }
        acc0 = __builtin_amdgcn_wmma_f32_16x16x32_f16(
                   false, afrag, false, bfrag0, (short)0, acc0, false, false);
        acc1 = __builtin_amdgcn_wmma_f32_16x16x32_f16(
                   false, afrag, false, bfrag1, (short)0, acc1, false, false);
        __syncthreads();
    }

    const int col0 = n0 + wc * 32 + r;
    const int col1 = col0 + 16;
    const float bv0 = bias ? bias[col0] : 0.0f;
    const float bv1 = bias ? bias[col1] : 0.0f;
#pragma unroll
    for (int v = 0; v < 8; ++v) {
        int row = m0 + wr * 16 + half * 8 + v;
        if (row < M) {
            float v0 = acc0[v] + bv0;
            float v1 = acc1[v] + bv1;
            if (OUT_F16) {
                Ch[(size_t)row * N + col0] = (_Float16)v0;
                Ch[(size_t)row * N + col1] = (_Float16)v1;
            } else {
                Cf[(size_t)row * N + col0] = v0;
                Cf[(size_t)row * N + col1] = v1;
            }
        }
    }
}

// ---------------- degree / norm ----------------
__global__ void deg_init(float* deg, int n) {
    int i = blockIdx.x * blockDim.x + threadIdx.x;
    if (i < n) deg[i] = 1.0f;   // self-loop
}
__global__ void deg_count(const int* __restrict__ dst, float* __restrict__ deg, int e) {
    int i = blockIdx.x * blockDim.x + threadIdx.x;
    if (i < e) atomAddF(&deg[dst[i]], 1.0f);
}
__global__ void deg_rsqrt(const float* __restrict__ deg, float* __restrict__ dinv, int n) {
    int i = blockIdx.x * blockDim.x + threadIdx.x;
    if (i < n) dinv[i] = rsqrtf(deg[i]);
}

// ---------------- edge scatter: wave per edge, contiguous row gather ----------------
__global__ __launch_bounds__(256) void edge_scatter(
    const float* __restrict__ h, float* __restrict__ agg,
    const int* __restrict__ src, const int* __restrict__ dst,
    const float* __restrict__ dinv, int E, int C)
{
    int wid  = (blockIdx.x * blockDim.x + threadIdx.x) >> 5;
    int lane = threadIdx.x & 31;
    if (wid >= E) return;
    int s = src[wid], d = dst[wid];
    float w = dinv[s] * dinv[d];
    const float* hrow = h + (size_t)s * C;
    float* arow = agg + (size_t)d * C;
    for (int c = lane; c < C; c += 32)
        atomAddF(&arow[c], hrow[c] * w);
}

// ---------------- self-loop term + bias ----------------
__global__ void selfloop_bias(const float* __restrict__ h, float* __restrict__ agg,
                              const float* __restrict__ dinv, const float* __restrict__ bias,
                              int n, int C)
{
    int i = blockIdx.x * blockDim.x + threadIdx.x;
    int tot = n * C;
    int stride = gridDim.x * blockDim.x;
    for (; i < tot; i += stride) {
        int row = i / C, c = i - row * C;
        float di = dinv[row];
        agg[i] += h[i] * di * di + bias[c];
    }
}

// ---------------- batch norm ----------------
__global__ __launch_bounds__(256) void bn_stats(const float* __restrict__ z,
                                                float* __restrict__ sums,
                                                float* __restrict__ sumsq, int n)
{
    int c  = threadIdx.x;          // channel (256)
    int r0 = blockIdx.x * 128;
    float s = 0.f, s2 = 0.f;
    int rend = r0 + 128; if (rend > n) rend = n;
    for (int r = r0; r < rend; ++r) {
        float v = z[(size_t)r * 256 + c];
        s += v; s2 += v * v;
    }
    atomAddF(&sums[c], s);
    atomAddF(&sumsq[c], s2);
}

__global__ void bn_finalize(const float* sums, const float* sumsq,
                            const float* gamma, const float* beta,
                            float* scale, float* shift, int n, int C)
{
    int c = blockIdx.x * blockDim.x + threadIdx.x;
    if (c >= C) return;
    float invn = 1.0f / (float)n;
    float mean = sums[c] * invn;
    float var  = sumsq[c] * invn - mean * mean;
    float sc   = gamma[c] * rsqrtf(var + BN_EPS);
    scale[c] = sc;
    shift[c] = beta[c] - mean * sc;
}

__global__ void bn_apply_relu(const float* __restrict__ z,
                              const float* __restrict__ scale,
                              const float* __restrict__ shift,
                              _Float16* __restrict__ out, int tot)
{
    int i = blockIdx.x * blockDim.x + threadIdx.x;
    int stride = gridDim.x * blockDim.x;
    for (; i < tot; i += stride) {
        int c = i & 255;
        float v = z[i] * scale[c] + shift[c];
        out[i] = (_Float16)fmaxf(v, 0.0f);
    }
}

// ---------------- host orchestration ----------------
extern "C" void kernel_launch(void* const* d_in, const int* in_sizes, int n_in,
                              void* d_out, int out_size, void* d_ws, size_t ws_size,
                              hipStream_t stream)
{
    const float* x       = (const float*)d_in[0];
    const float* y       = (const float*)d_in[1];
    const int*   ei      = (const int*)d_in[2];
    const float* fc1_w   = (const float*)d_in[3];
    const float* fc1_b   = (const float*)d_in[4];
    const float* fc2_w   = (const float*)d_in[5];
    const float* fc2_b   = (const float*)d_in[6];
    const float* conv1_w = (const float*)d_in[7];
    const float* conv1_b = (const float*)d_in[8];
    const float* conv2_w = (const float*)d_in[9];
    const float* conv2_b = (const float*)d_in[10];
    const float* gamma   = (const float*)d_in[11];
    const float* beta    = (const float*)d_in[12];
    const int* src = ei;
    const int* dst = ei + N_EDGE;

    // carve workspace
    char* p = (char*)d_ws;
    auto carve = [&](size_t bytes) -> void* {
        void* r = (void*)p;
        p += (bytes + 255) & ~(size_t)255;
        return r;
    };
    _Float16* xh  = (_Float16*)carve((size_t)N_USR * F_USR * 2);
    _Float16* yh  = (_Float16*)carve((size_t)N_MOV * F_MOV * 2);
    _Float16* w1h = (_Float16*)carve((size_t)F_USR * INC * 2);
    _Float16* w2h = (_Float16*)carve((size_t)F_MOV * INC * 2);
    _Float16* c1h = (_Float16*)carve((size_t)INC * HIDC * 2);
    _Float16* c2h = (_Float16*)carve((size_t)HIDC * OUTC * 2);
    _Float16* z1h = (_Float16*)carve((size_t)N_TOT * INC * 2);
    float*    h1   = (float*)carve((size_t)N_TOT * HIDC * 4);
    float*    agg1 = (float*)carve((size_t)N_TOT * HIDC * 4);
    _Float16* z2h  = (_Float16*)carve((size_t)N_TOT * HIDC * 2);
    float*    h2   = (float*)carve((size_t)N_TOT * OUTC * 4);
    float*    deg  = (float*)carve((size_t)N_TOT * 4);
    float*    dinv = (float*)carve((size_t)N_TOT * 4);
    float*    sums  = (float*)carve(256 * 4);
    float*    sumsq = (float*)carve(256 * 4);
    float*    scale = (float*)carve(256 * 4);
    float*    shift = (float*)carve(256 * 4);
    float*    outF  = (float*)d_out;

    // zero accumulators every call (graph-capture safe)
    (void)hipMemsetAsync(agg1,  0, (size_t)N_TOT * HIDC * 4, stream);
    (void)hipMemsetAsync(d_out, 0, (size_t)N_TOT * OUTC * 4, stream);
    (void)hipMemsetAsync(sums,  0, 256 * 4, stream);
    (void)hipMemsetAsync(sumsq, 0, 256 * 4, stream);

    // conversions to f16
    auto cvt = [&](const float* in, _Float16* out, int n) {
        int blocks = (n + 255) / 256;
        cvt_f32_f16<<<blocks, 256, 0, stream>>>(in, out, n);
    };
    cvt(x, xh, N_USR * F_USR);
    cvt(y, yh, N_MOV * F_MOV);
    cvt(fc1_w, w1h, F_USR * INC);
    cvt(fc2_w, w2h, F_MOV * INC);
    cvt(conv1_w, c1h, INC * HIDC);
    cvt(conv2_w, c2h, HIDC * OUTC);

    // degrees / symmetric norm
    deg_init<<<(N_TOT + 255) / 256, 256, 0, stream>>>(deg, N_TOT);
    deg_count<<<(N_EDGE + 255) / 256, 256, 0, stream>>>(dst, deg, N_EDGE);
    deg_rsqrt<<<(N_TOT + 255) / 256, 256, 0, stream>>>(deg, dinv, N_TOT);

    // fc1 / fc2 -> z1 (f16, concatenated)
    {
        dim3 g1((N_USR + 63) / 64, INC / 64);
        wmma_gemm<true><<<g1, 256, 0, stream>>>(xh, w1h, fc1_b, nullptr, z1h,
                                                N_USR, INC, F_USR);
        dim3 g2((N_MOV + 63) / 64, INC / 64);
        wmma_gemm<true><<<g2, 256, 0, stream>>>(yh, w2h, fc2_b, nullptr,
                                                z1h + (size_t)N_USR * INC,
                                                N_MOV, INC, F_MOV);
    }

    // conv1: h1 = z1 @ conv1_w (bias after aggregation)
    {
        dim3 g((N_TOT + 63) / 64, HIDC / 64);
        wmma_gemm<false><<<g, 256, 0, stream>>>(z1h, c1h, nullptr, h1, nullptr,
                                                N_TOT, HIDC, INC);
    }
    edge_scatter<<<(N_EDGE * 32 + 255) / 256, 256, 0, stream>>>(h1, agg1, src, dst,
                                                                dinv, N_EDGE, HIDC);
    selfloop_bias<<<(N_TOT * HIDC + 255) / 256, 256, 0, stream>>>(h1, agg1, dinv,
                                                                  conv1_b, N_TOT, HIDC);

    // batch norm + relu -> z2 (f16)
    bn_stats<<<(N_TOT + 127) / 128, 256, 0, stream>>>(agg1, sums, sumsq, N_TOT);
    bn_finalize<<<1, 256, 0, stream>>>(sums, sumsq, gamma, beta, scale, shift, N_TOT, HIDC);
    bn_apply_relu<<<(N_TOT * HIDC + 255) / 256, 256, 0, stream>>>(agg1, scale, shift,
                                                                  z2h, N_TOT * HIDC);

    // conv2: h2 = z2 @ conv2_w; aggregate into d_out
    {
        dim3 g((N_TOT + 63) / 64, OUTC / 64);
        wmma_gemm<false><<<g, 256, 0, stream>>>(z2h, c2h, nullptr, h2, nullptr,
                                                N_TOT, OUTC, HIDC);
    }
    edge_scatter<<<(N_EDGE * 32 + 255) / 256, 256, 0, stream>>>(h2, outF, src, dst,
                                                                dinv, N_EDGE, OUTC);
    selfloop_bias<<<(N_TOT * OUTC + 255) / 256, 256, 0, stream>>>(h2, outF, dinv,
                                                                  conv2_b, N_TOT, OUTC);
}
